// MultiHeadedAttentionWrapper_24120536335055
// MI455X (gfx1250) — matrix operand order repaired and checked by reading, not compiled, not run on hardware
//
#include <hip/hip_runtime.h>
#include <hip/hip_bf16.h>
#include <stdint.h>

typedef __attribute__((ext_vector_type(16))) _Float16 v16h;
typedef __attribute__((ext_vector_type(8)))  float    v8f;

union AB { uint32_t u[8]; uint4 q[2]; v16h v; };
union CD { float    f[8]; v8f  v; };

#define BATCH 8
#define SEQ   1024
#define DIN   768
#define NH    12
#define DK    64
#define WPN   (NH * (DIN/2) * DK)      // u32 words per weight matrix = 294912
#define XHN   (BATCH * SEQ * (DIN/2))  // u32 words of packed x     = 3145728

static __device__ __forceinline__ uint32_t pack2h(float a, float b) {
    _Float16 ha = (_Float16)a, hb = (_Float16)b;
    uint16_t ua = __builtin_bit_cast(uint16_t, ha);
    uint16_t ub = __builtin_bit_cast(uint16_t, hb);
    return (uint32_t)ua | ((uint32_t)ub << 16);
}

static __device__ __forceinline__ v8f wmma_f16(const AB& a, const AB& b, v8f c) {
    return __builtin_amdgcn_wmma_f32_16x16x32_f16(false, a.v, false, b.v,
                                                  (short)0, c, false, false);
}

// ---------------------------------------------------------------------------
// Kernel 0: pack x f32 -> f16 pairs (row-major, pairs along d_in are already
// contiguous, so the GEMM A-operand becomes two b128 loads per chunk).
// ---------------------------------------------------------------------------
__global__ void mha_pack_x(const float* __restrict__ x, uint32_t* __restrict__ xh) {
    int idx = blockIdx.x * blockDim.x + threadIdx.x;
    if (idx >= XHN) return;
    float2 v = ((const float2*)x)[idx];
    xh[idx] = pack2h(v.x, v.y);
}

// ---------------------------------------------------------------------------
// Kernel 1: pack weights into per-lane-contiguous "B-tile" order:
//   WpT[mat][h][kc][halfHi][n][v],  word = pack(W[h][2kp][n], W[h][2kp+1][n])
//   with kp = kc*16 + halfHi*8 + v.
// A lane (column n, half halfHi) then reads its 8 B-operand dwords for chunk
// kc as one contiguous 32 B run -> 2 x global_load_b128.
// ---------------------------------------------------------------------------
__global__ void mha_pack_w(const float* __restrict__ Wq,
                           const float* __restrict__ Wk,
                           const float* __restrict__ Wv,
                           uint32_t* __restrict__ WpT) {
    int idx = blockIdx.x * blockDim.x + threadIdx.x;
    if (idx >= 3 * WPN) return;
    int mat = idx / WPN;
    int r   = idx % WPN;
    const float* W = (mat == 0) ? Wq : (mat == 1) ? Wk : Wv;
    int v      = r & 7;   r >>= 3;
    int n      = r & 63;  r >>= 6;
    int halfHi = r & 1;   r >>= 1;
    int kc     = r % 24;
    int h      = r / 24;
    int kp = kc * 16 + halfHi * 8 + v;
    float lo = W[((size_t)h * DIN + 2*kp    ) * DK + n];
    float hi = W[((size_t)h * DIN + 2*kp + 1) * DK + n];
    WpT[idx] = pack2h(lo, hi);
}

// ---------------------------------------------------------------------------
// Kernel 2: QKV projection.  Block = 4 waves; wave owns a 16-wide output slab
// (n0 = wave*16).  24 K-chunks of 32 over DIN=768, 3 WMMAs per chunk.
//   Q f16 [bh][s][64] pre-scaled by 0.125;  K f16 [bh][s][64];
//   V f16 transposed [bh][64][1024] (stored as one b128 per lane).
// ---------------------------------------------------------------------------
__global__ __launch_bounds__(128)
void mha_qkv(const uint32_t* __restrict__ xh, const uint32_t* __restrict__ WpT,
             _Float16* __restrict__ Qh, _Float16* __restrict__ Kh,
             _Float16* __restrict__ Vt) {
    const int lane   = threadIdx.x & 31;
    const int wave   = threadIdx.x >> 5;
    const int halfHi = lane >> 4;
    const int nl     = lane & 15;

    int blk = blockIdx.x;              // b*NH*64 + h*64 + st
    int st  = blk % 64;
    int bh  = blk / 64;
    int h   = bh % NH;
    int b   = bh / NH;
    int s0  = st * 16;
    int n0  = wave * 16;

    CD cq, ck, cv;
    #pragma unroll
    for (int i = 0; i < 8; ++i) { cq.f[i] = 0.f; ck.f[i] = 0.f; cv.f[i] = 0.f; }

    // A-operand: row m = nl (both lane halves), K-pairs contiguous in xh.
    const uint32_t* xp = xh + (size_t)(b * SEQ + s0 + nl) * (DIN/2);

    for (int kc = 0; kc < 24; ++kc) {
        AB a;
        a.q[0] = *(const uint4*)(xp + kc*16 + halfHi*8);
        a.q[1] = *(const uint4*)(xp + kc*16 + halfHi*8 + 4);

        const uint32_t* wb = WpT +
            (((((size_t)h * 24 + kc) * 2 + halfHi) * 64) + (n0 + nl)) * 8;
        AB bq, bk, bv;
        bq.q[0] = *(const uint4*)(wb);
        bq.q[1] = *(const uint4*)(wb + 4);
        bk.q[0] = *(const uint4*)(wb + (size_t)WPN);
        bk.q[1] = *(const uint4*)(wb + (size_t)WPN + 4);
        bv.q[0] = *(const uint4*)(wb + 2*(size_t)WPN);
        bv.q[1] = *(const uint4*)(wb + 2*(size_t)WPN + 4);

        cq.v = wmma_f16(a, bq, cq.v);
        ck.v = wmma_f16(a, bk, ck.v);
        cv.v = wmma_f16(a, bv, cv.v);
    }

    // C/D layout: VGPR i -> row = i + 8*halfHi, col = nl
    size_t base = (size_t)(b * NH + h) * SEQ * DK;
    int col = n0 + nl;
    #pragma unroll
    for (int i = 0; i < 8; ++i) {
        int row = s0 + i + 8 * halfHi;
        Qh[base + (size_t)row * DK + col] = (_Float16)(cq.f[i] * 0.125f);
        Kh[base + (size_t)row * DK + col] = (_Float16)ck.f[i];
    }
    // V transposed: rows are consecutive along i -> one 16B store of 8 f16.
    uint4 vpack;
    vpack.x = pack2h(cv.f[0], cv.f[1]);
    vpack.y = pack2h(cv.f[2], cv.f[3]);
    vpack.z = pack2h(cv.f[4], cv.f[5]);
    vpack.w = pack2h(cv.f[6], cv.f[7]);
    *(uint4*)(Vt + base + (size_t)col * SEQ + s0 + 8 * halfHi) = vpack;
}

// ---------------------------------------------------------------------------
// Kernel 3: flash attention.  One wave per 16-query tile; 32 keys/iteration.
//   S:  4 WMMAs  (2 key subtiles x K=64 contraction)
//   PV: 4 WMMAs  (4 feature slabs x K=32 keys)
// All WMMA operands are 2 x b128 per lane; P staged via wave-private LDS.
// ---------------------------------------------------------------------------
__global__ __launch_bounds__(128)
void mha_attn(const _Float16* __restrict__ Qh, const _Float16* __restrict__ Kh,
              const _Float16* __restrict__ Vt, float* __restrict__ out) {
    __shared__ _Float16 plds[4][16 * 32];  // 1 KB per wave

    const int lane   = threadIdx.x & 31;
    const int wave   = threadIdx.x >> 5;
    const int halfHi = lane >> 4;
    const int nl     = lane & 15;

    int tid = blockIdx.x * 4 + wave;
    int qi  = tid & 63;
    int bh  = tid >> 6;
    int h   = bh % NH;
    int b   = bh / NH;
    int q0  = qi * 16;

    size_t base = (size_t)(b * NH + h) * SEQ * DK;
    const uint32_t* Q = (const uint32_t*)(Qh + base);  // 32 dwords per row
    const uint32_t* K = (const uint32_t*)(Kh + base);
    const uint32_t* V = (const uint32_t*)(Vt + base);  // [64][512 dwords]

    // Q tile in A layout: two K=32 chunks over d_head=64.
    AB qa[2];
    {
        const uint32_t* qrow = Q + (size_t)(q0 + nl) * 32;
        #pragma unroll
        for (int c = 0; c < 2; ++c) {
            qa[c].q[0] = *(const uint4*)(qrow + c*16 + halfHi*8);
            qa[c].q[1] = *(const uint4*)(qrow + c*16 + halfHi*8 + 4);
        }
    }

    float mrow[8], srow[8];
    CD o[4];
    #pragma unroll
    for (int i = 0; i < 8; ++i) {
        mrow[i] = -INFINITY; srow[i] = 0.f;
        o[0].f[i] = 0.f; o[1].f[i] = 0.f; o[2].f[i] = 0.f; o[3].f[i] = 0.f;
    }

    _Float16* pl = plds[wave];
    const int nIter = (q0 + 16 + 31) / 32;

    for (int it = 0; it < nIter; ++it) {
        int kb = it * 32;

        if (it + 1 < nIter) {  // global_prefetch of next K rows
            __builtin_prefetch(K + (size_t)(kb + 32 + nl) * 32, 0, 0);
        }

        // ---- S = Q * K^T ----
        CD sc[2];
        #pragma unroll
        for (int sub = 0; sub < 2; ++sub) {
            const uint32_t* krow = K + (size_t)(kb + sub*16 + nl) * 32 + halfHi*8;
            AB k0, k1;
            k0.q[0] = *(const uint4*)(krow);
            k0.q[1] = *(const uint4*)(krow + 4);
            k1.q[0] = *(const uint4*)(krow + 16);
            k1.q[1] = *(const uint4*)(krow + 20);
            CD z;
            #pragma unroll
            for (int i = 0; i < 8; ++i) z.f[i] = 0.f;
            v8f acc = wmma_f16(qa[0], k0, z.v);
            acc     = wmma_f16(qa[1], k1, acc);
            sc[sub].v = acc;
        }

        // ---- causal mask + online softmax ----
        #pragma unroll
        for (int i = 0; i < 8; ++i) {
            int row = q0 + i + 8 * halfHi;
            float s0v = (kb + nl      <= row) ? sc[0].f[i] : -INFINITY;
            float s1v = (kb + 16 + nl <= row) ? sc[1].f[i] : -INFINITY;

            float mx = fmaxf(s0v, s1v);
            mx = fmaxf(mx, __shfl_xor(mx, 1, 32));
            mx = fmaxf(mx, __shfl_xor(mx, 2, 32));
            mx = fmaxf(mx, __shfl_xor(mx, 4, 32));
            mx = fmaxf(mx, __shfl_xor(mx, 8, 32));

            float mnew  = fmaxf(mrow[i], mx);        // finite: key kb <= row
            float alpha = __expf(mrow[i] - mnew);
            mrow[i] = mnew;

            float p0 = __expf(s0v - mnew);
            float p1 = __expf(s1v - mnew);

            float rs = p0 + p1;
            rs += __shfl_xor(rs, 1, 32);
            rs += __shfl_xor(rs, 2, 32);
            rs += __shfl_xor(rs, 4, 32);
            rs += __shfl_xor(rs, 8, 32);
            srow[i] = srow[i] * alpha + rs;

            int lrow = i + 8 * halfHi;
            pl[lrow * 32 + nl]      = (_Float16)p0;
            pl[lrow * 32 + 16 + nl] = (_Float16)p1;

            o[0].f[i] *= alpha; o[1].f[i] *= alpha;
            o[2].f[i] *= alpha; o[3].f[i] *= alpha;
        }

        // Wave-private LDS region: order store->load within the wave.
        asm volatile("s_wait_dscnt 0" ::: "memory");

        // ---- P back in A layout (two ds_load_b128) ----
        AB pa;
        {
            const uint32_t* pl32 = (const uint32_t*)pl + nl*16 + halfHi*8;
            pa.q[0] = *(const uint4*)(pl32);
            pa.q[1] = *(const uint4*)(pl32 + 4);
        }

        // ---- O += P * V ----
        #pragma unroll
        for (int j = 0; j < 4; ++j) {
            const uint32_t* vrow = V + (size_t)(j*16 + nl) * 512 + kb/2 + halfHi*8;
            AB vb;
            vb.q[0] = *(const uint4*)(vrow);
            vb.q[1] = *(const uint4*)(vrow + 4);
            o[j].v = wmma_f16(pa, vb, o[j].v);
        }
    }

    // ---- normalize and store ----
    #pragma unroll
    for (int i = 0; i < 8; ++i) {
        int row = q0 + i + 8 * halfHi;
        float inv = 1.0f / srow[i];
        #pragma unroll
        for (int j = 0; j < 4; ++j) {
            out[((size_t)(b * SEQ) + row) * (NH * DK) + h * DK + j * 16 + nl]
                = o[j].f[i] * inv;
        }
    }
}

// ---------------------------------------------------------------------------
// Workspace layout (bytes), all 16B-aligned:
//   xh  : XHN*4              = 12,582,912
//   WpT : 3*WPN*4            =  3,538,944
//   Qh  : 8*12*1024*64*2     = 12,582,912
//   Kh  : 12,582,912
//   Vt  : 12,582,912
//   total ~= 53.9 MB
// ---------------------------------------------------------------------------
extern "C" void kernel_launch(void* const* d_in, const int* in_sizes, int n_in,
                              void* d_out, int out_size, void* d_ws, size_t ws_size,
                              hipStream_t stream) {
    const float* x  = (const float*)d_in[0];
    const float* Wq = (const float*)d_in[1];
    const float* Wk = (const float*)d_in[2];
    const float* Wv = (const float*)d_in[3];
    float* out = (float*)d_out;

    uint8_t* ws = (uint8_t*)d_ws;
    const size_t xhBytes  = (size_t)XHN * 4;                    // 12,582,912
    const size_t wpBytes  = (size_t)3 * WPN * 4;                //  3,538,944
    const size_t qkvBytes = (size_t)BATCH * NH * SEQ * DK * 2;  // 12,582,912
    uint32_t* xh  = (uint32_t*)ws;
    uint32_t* WpT = (uint32_t*)(ws + xhBytes);
    _Float16* Qh  = (_Float16*)(ws + xhBytes + wpBytes);
    _Float16* Kh  = (_Float16*)(ws + xhBytes + wpBytes + qkvBytes);
    _Float16* Vt  = (_Float16*)(ws + xhBytes + wpBytes + 2 * qkvBytes);

    mha_pack_x<<<(XHN + 255) / 256, 256, 0, stream>>>(x, xh);
    mha_pack_w<<<(3 * WPN + 255) / 256, 256, 0, stream>>>(Wq, Wk, Wv, WpT);
    mha_qkv<<<BATCH * NH * (SEQ / 16), 128, 0, stream>>>(xh, WpT, Qh, Kh, Vt);
    mha_attn<<<BATCH * NH * (SEQ / 16) / 4, 128, 0, stream>>>(Qh, Kh, Vt, out);
}
